// EMAVectorQuantizer_64493228916940
// MI455X (gfx1250) — matrix operand order repaired and checked by reading, not compiled, not run on hardware
//
#include <hip/hip_runtime.h>
#include <hip/hip_bf16.h>

typedef __attribute__((ext_vector_type(16))) __bf16 v16bf;
typedef __attribute__((ext_vector_type(8)))  float  v8f;

#define VQ_D       256
#define VQ_K       8192
#define VQ_N       32768
#define KTILES     (VQ_K / 16)        // 512
#define DCHUNKS    (VQ_D / 32)        // 8
#define FRAG       (DCHUNKS * 32 * 16)  // 4096 bf16 = 8KB per k-tile per array
#define TPS        2                  // k-tiles per LDS stage
#define STAGE      (FRAG * TPS)       // 8192 bf16 = 16KB per array per stage
#define NSTAGES    (KTILES / TPS)     // 256
#define PACK_ELEMS (KTILES * FRAG)    // 2,097,152 bf16 per array

// gfx1250 async global->LDS copy path (ASYNCcnt-tracked), with a safe
// register-pipelined fallback if the builtin is absent on this toolchain.
#if __has_builtin(__builtin_amdgcn_global_load_async_to_lds_b128)
#define VQ_ASYNC 1
#else
#define VQ_ASYNC 0
#endif

#if VQ_ASYNC
#if __has_builtin(__builtin_amdgcn_s_wait_asynccnt)
#define VQ_WAIT_ASYNC(n) __builtin_amdgcn_s_wait_asynccnt(n)
#else
#define VQ_WAIT_ASYNC(n) asm volatile("s_wait_asynccnt %0" ::"n"(n))
#endif
// Builtin parameter types: pointers to the NATIVE 16-byte int vector
// (HIP's uint4 is a wrapper class), non-const, AS(1) src / AS(3) dst.
typedef int vq_b128 __attribute__((vector_size(16)));
typedef __attribute__((address_space(1))) vq_b128 gb128;
typedef __attribute__((address_space(3))) vq_b128 lb128;
#endif

// ---------------------------------------------------------------------------
// Prep 1: pack embed[D,K] (f32) into bf16 hi/lo arrays in WMMA B-fragment
// order: index = (((T*8 + c)*32 + lane)*16 + e)
//   column n = lane&15 -> k = T*16 + n ; contraction d = c*32 + (lane>>4)*16 + e
// so each GEMM lane reads its 16 bf16 (32B) contiguously.
// ---------------------------------------------------------------------------
__global__ void vq_pack_b(const float* __restrict__ embed,
                          __bf16* __restrict__ packHi,
                          __bf16* __restrict__ packLo) {
  int i = blockIdx.x * blockDim.x + threadIdx.x;
  int e = i & 15;
  int l = (i >> 4) & 31;
  int c = (i >> 9) & 7;
  int T = i >> 12;
  int d = c * 32 + (l >> 4) * 16 + e;
  int k = T * 16 + (l & 15);
  float f = embed[d * VQ_K + k];
  __bf16 h  = (__bf16)f;
  __bf16 lo = (__bf16)(f - (float)h);
  packHi[i] = h;
  packLo[i] = lo;
}

// ---------------------------------------------------------------------------
// Prep 2: bias[k] = 0.5 * ||e_k||^2 ; also zero the loss slot each call
// (harness poisons d_out; graph replays must be deterministic).
// ---------------------------------------------------------------------------
__global__ void vq_bias(const float* __restrict__ embed,
                        float* __restrict__ bias,
                        float* __restrict__ lossSlot) {
  int k = blockIdx.x * blockDim.x + threadIdx.x;
  float s = 0.0f;
  for (int d = 0; d < VQ_D; ++d) {
    float v = embed[d * VQ_K + k];
    s += v * v;
  }
  bias[k] = 0.5f * s;
  if (k == 0) *lossSlot = 0.0f;
}

// ---------------------------------------------------------------------------
// Prep 3: embedT[k][d] = embed[d][k]  (makes the final gather contiguous)
// ---------------------------------------------------------------------------
__global__ void vq_transpose(const float* __restrict__ embed,
                             float* __restrict__ embedT) {
  int i = blockIdx.x * blockDim.x + threadIdx.x;  // i = k*256 + d
  int k = i >> 8;
  int d = i & 255;
  embedT[i] = embed[d * VQ_K + k];
}

#if VQ_ASYNC
// Issue one stage (2 k-tiles, 16KB hi + 16KB lo) as 8 async b128 copies per
// thread, direct global->LDS (no VGPR round trip), tracked by ASYNCcnt.
__device__ __forceinline__ void vq_stage_async(const __bf16* gHi,
                                               const __bf16* gLo,
                                               __bf16* sHi, __bf16* sLo,
                                               int tid) {
  gb128* gh = (gb128*)gHi + tid;
  gb128* gl = (gb128*)gLo + tid;
  lb128* sh = (lb128*)sHi + tid;
  lb128* sl = (lb128*)sLo + tid;
  #pragma unroll
  for (int j = 0; j < 4; ++j) {
    __builtin_amdgcn_global_load_async_to_lds_b128(gh + j * 256, sh + j * 256, 0, 0);
    __builtin_amdgcn_global_load_async_to_lds_b128(gl + j * 256, sl + j * 256, 0, 0);
  }
}
#endif

// ---------------------------------------------------------------------------
// Main kernel: fused  score = x.e - 0.5||e||^2  GEMM (bf16-split WMMA)
// + running argmax over all K.  256 threads = 8 wave32; wave w owns rows
// [blk*128 + w*16, +16). 256 stages of 2 k-tiles; B double-buffered in LDS
// with async prefetch so the 48 WMMAs per stage overlap the next copy.
// ---------------------------------------------------------------------------
__launch_bounds__(256)
__global__ void vq_argmax(const float* __restrict__ x,
                          const __bf16* __restrict__ packHi,
                          const __bf16* __restrict__ packLo,
                          const float* __restrict__ bias,
                          int* __restrict__ idxOut,
                          float* __restrict__ idxOutF) {
#if VQ_ASYNC
  __shared__ __align__(32) __bf16 ldsHi[2][STAGE];   // 2 x 16KB
  __shared__ __align__(32) __bf16 ldsLo[2][STAGE];   // 2 x 16KB
#else
  __shared__ __align__(32) __bf16 ldsHi[1][STAGE];   // 16KB
  __shared__ __align__(32) __bf16 ldsLo[1][STAGE];   // 16KB
#endif

  const int lane    = threadIdx.x & 31;
  const int wave    = threadIdx.x >> 5;
  const int rowBase = blockIdx.x * 128 + wave * 16;
  const int m       = lane & 15;   // A-row / B-column owned by this lane
  const int khalf   = lane >> 4;

  // --- Load the wave's 16x256 fp32 A strip once; split to bf16 hi/lo in the
  //     contiguous k = 16*khalf + e order, as in packed B (VGPR-resident across all k-tiles).
  v16bf a_hi[DCHUNKS], a_lo[DCHUNKS];
  const float* xr = x + (rowBase + m) * VQ_D;
  #pragma unroll
  for (int c = 0; c < DCHUNKS; ++c) {
    const int d0 = c * 32 + khalf * 16;
    #pragma unroll
    for (int e = 0; e < 16; ++e) {
      // K = d0+e (contiguous, matches packed B)
      float f = xr[d0 + e];
      __bf16 h = (__bf16)f;
      a_hi[c][e] = h;
      a_lo[c][e] = (__bf16)(f - (float)h);
    }
  }

  float bestVal[8];
  int   bestIdx[8];
  #pragma unroll
  for (int r = 0; r < 8; ++r) { bestVal[r] = -3.0e38f; bestIdx[r] = 0; }

#if VQ_ASYNC
  // Prologue: kick off stage 0 into buffer 0.
  vq_stage_async(packHi, packLo, &ldsHi[0][0], &ldsLo[0][0], threadIdx.x);
#else
  // Fallback software pipeline: next stage staged through 32 VGPRs.
  uint4 rH[4], rL[4];
  {
    const uint4* gh = (const uint4*)packHi + threadIdx.x;
    const uint4* gl = (const uint4*)packLo + threadIdx.x;
    #pragma unroll
    for (int j = 0; j < 4; ++j) { rH[j] = gh[j * 256]; rL[j] = gl[j * 256]; }
  }
#endif

  for (int s = 0; s < NSTAGES; ++s) {
#if VQ_ASYNC
    const int cur = s & 1;
    if (s + 1 < NSTAGES) {
      // Prefetch stage s+1 into the other buffer, then drain only stage s's
      // 8 copies (async loads complete in order): overlap with 48 WMMAs.
      vq_stage_async(packHi + (size_t)(s + 1) * STAGE,
                     packLo + (size_t)(s + 1) * STAGE,
                     &ldsHi[cur ^ 1][0], &ldsLo[cur ^ 1][0], threadIdx.x);
      VQ_WAIT_ASYNC(8);
    } else {
      VQ_WAIT_ASYNC(0);
    }
    __syncthreads();   // all waves' stage-s copies are in LDS
#else
    const int cur = 0;
    __syncthreads();   // previous stage's readers done
    {
      uint4* sh = (uint4*)&ldsHi[0][0] + threadIdx.x;
      uint4* sl = (uint4*)&ldsLo[0][0] + threadIdx.x;
      #pragma unroll
      for (int j = 0; j < 4; ++j) { sh[j * 256] = rH[j]; sl[j * 256] = rL[j]; }
    }
    __syncthreads();
    if (s + 1 < NSTAGES) {  // issue next-stage loads; latency hidden by WMMAs
      const uint4* gh = (const uint4*)(packHi + (size_t)(s + 1) * STAGE) + threadIdx.x;
      const uint4* gl = (const uint4*)(packLo + (size_t)(s + 1) * STAGE) + threadIdx.x;
      #pragma unroll
      for (int j = 0; j < 4; ++j) { rH[j] = gh[j * 256]; rL[j] = gl[j * 256]; }
    }
#endif

    #pragma unroll
    for (int t = 0; t < TPS; ++t) {
      const __bf16* bHiT = &ldsHi[cur][t * FRAG];
      const __bf16* bLoT = &ldsLo[cur][t * FRAG];
      v8f acc = {};
      #pragma unroll
      for (int c = 0; c < DCHUNKS; ++c) {
        v16bf bh = *(const v16bf*)(bHiT + (c * 32 + lane) * 16);
        v16bf bl = *(const v16bf*)(bLoT + (c * 32 + lane) * 16);
        // fp32-accurate split: hi*hi + lo*hi + hi*lo (lo*lo ~ 2^-32, dropped)
        acc = __builtin_amdgcn_wmma_f32_16x16x32_bf16(false, a_hi[c], false, bh,
                                                      (short)0, acc, false, false);
        acc = __builtin_amdgcn_wmma_f32_16x16x32_bf16(false, a_lo[c], false, bh,
                                                      (short)0, acc, false, false);
        acc = __builtin_amdgcn_wmma_f32_16x16x32_bf16(false, a_hi[c], false, bl,
                                                      (short)0, acc, false, false);
      }
      // Epilogue: score = acc - 0.5||e_k||^2 ; running argmax.
      // C/D layout: reg r, lane -> (row = r + 8*khalf, col k = tile*16 + m)
      const int kcol = (s * TPS + t) * 16 + m;
      const float b = bias[kcol];
      #pragma unroll
      for (int r = 0; r < 8; ++r) {
        float sc = acc[r] - b;
        if (sc > bestVal[r]) { bestVal[r] = sc; bestIdx[r] = kcol; }
      }
    }
#if VQ_ASYNC
    __syncthreads();   // readers done before stage s+2 overwrites this buffer
#endif
  }

  // Cross-lane argmax over the 16 lanes sharing each output row
  // (xor offsets <= 8 stay within each 16-lane half). Ties -> lowest index,
  // matching jnp.argmin semantics.
  #pragma unroll
  for (int r = 0; r < 8; ++r) {
    float v  = bestVal[r];
    int   bi = bestIdx[r];
    #pragma unroll
    for (int off = 8; off >= 1; off >>= 1) {
      float ov = __shfl_xor(v, off, 32);
      int   oi = __shfl_xor(bi, off, 32);
      if (ov > v || (ov == v && oi < bi)) { v = ov; bi = oi; }
    }
    if (m == 0) {
      int row = rowBase + khalf * 8 + r;
      idxOut[row]  = bi;
      idxOutF[row] = (float)bi;   // indices region of d_out is fp32
    }
  }
}

// ---------------------------------------------------------------------------
// Gather + straight-through output + commitment loss.
// One block per token row: contiguous 1KB read of embedT[idx], fused MSE.
// ---------------------------------------------------------------------------
__launch_bounds__(256)
__global__ void vq_gather_loss(const float* __restrict__ x,
                               const float* __restrict__ embedT,
                               const int* __restrict__ idx,
                               float* __restrict__ outQ,
                               float* __restrict__ lossSlot) {
  const int n = blockIdx.x;
  const int d = threadIdx.x;
  const int k = idx[n];
  float xi = x[n * VQ_D + d];
  float q  = embedT[k * VQ_D + d];
  // straight-through estimator, computed exactly as the reference does
  outQ[n * VQ_D + d] = xi + (q - xi);
  float diff = q - xi;
  float p = diff * diff;

  // block reduction (wave32 tree + 8-wave LDS combine)
  __shared__ float red[8];
  #pragma unroll
  for (int off = 16; off >= 1; off >>= 1) p += __shfl_down(p, off, 32);
  if ((threadIdx.x & 31) == 0) red[threadIdx.x >> 5] = p;
  __syncthreads();
  if (threadIdx.x < 8) {
    float s = red[threadIdx.x];
    #pragma unroll
    for (int off = 4; off >= 1; off >>= 1) s += __shfl_down(s, off, 32);
    if (threadIdx.x == 0) {
      const float scale = 0.25f / (float)((long long)VQ_N * VQ_D);
      atomicAdd(lossSlot, s * scale);
    }
  }
}

// ---------------------------------------------------------------------------
// d_out layout (reference return order, fp32): quantized_st [N*D] | loss [1]
// | encoding_indices [N].
// ---------------------------------------------------------------------------
extern "C" void kernel_launch(void* const* d_in, const int* in_sizes, int n_in,
                              void* d_out, int out_size, void* d_ws, size_t ws_size,
                              hipStream_t stream) {
  const float* x     = (const float*)d_in[0];   // [N, D]
  const float* embed = (const float*)d_in[1];   // [D, K]

  float* outQ     = (float*)d_out;
  float* lossSlot = outQ + (size_t)VQ_N * VQ_D;
  float* idxF     = lossSlot + 1;

  // workspace carve-up (~16.2 MB)
  char* ws = (char*)d_ws;
  __bf16* packHi = (__bf16*)ws;                               // 4 MB
  __bf16* packLo = (__bf16*)(ws + (size_t)PACK_ELEMS * 2);    // 4 MB
  float*  bias   = (float*)(ws + (size_t)PACK_ELEMS * 4);     // 32 KB
  float*  embedT = bias + VQ_K;                               // 8 MB
  int*    idxI   = (int*)(embedT + (size_t)VQ_K * VQ_D);      // 128 KB

  vq_pack_b     <<<PACK_ELEMS / 256, 256, 0, stream>>>(embed, packHi, packLo);
  vq_bias       <<<VQ_K / 256,       256, 0, stream>>>(embed, bias, lossSlot);
  vq_transpose  <<<(VQ_K * VQ_D) / 256, 256, 0, stream>>>(embed, embedT);
  vq_argmax     <<<VQ_N / 128,       256, 0, stream>>>(x, packHi, packLo, bias,
                                                       idxI, idxF);
  vq_gather_loss<<<VQ_N,             256, 0, stream>>>(x, embedT, idxI, outQ,
                                                       lossSlot);
}
